// DDPM_40389872451638
// MI455X (gfx1250) — compile-verified, matrix operand order checked by reading, not run on hardware
//
#include <hip/hip_runtime.h>
#include <hip/hip_bf16.h>

// MI455X / gfx1250, wave32. Fused DDPM dynamic-conv pipeline, round 4:
//  - prep: pre-swizzle gen_w/fuse_w into bf16 A-fragment order + packed LUTs.
//  - K1: k = gen_w@y (+bias folded into WMMA C via float4 LDS loads) with
//    v_wmma_f32_16x16x32_bf16; dilated depthwise scatter is fully LDS-resident
//    and split into a batched gather clause (8x ds_load_u16, one wait) followed
//    by 8x ds_add_f32 (branch-free, no global traffic in the loop).
//  - K2: 3x3 conv [256->64] implicit-im2col WMMA; cat tile in LDS (bf16),
//    32 px/block, one A fragment + one batched LUT clause feed two WMMAs.

typedef __attribute__((ext_vector_type(16))) __bf16          v16bf;
typedef __attribute__((ext_vector_type(8)))  float           v8f;
typedef __attribute__((ext_vector_type(16))) unsigned short  v16us;

constexpr int N_ = 4, C_ = 64, H_ = 128, W_ = 128;
constexpr int KS    = 3;
constexpr int ROWS  = 3 * C_ * KS * KS;     // 1728 generated-kernel rows
constexpr int RT    = ROWS / 16;            // 108 row tiles
constexpr int PIX_T = 16;                   // pixels per K1 block
constexpr int WPB   = 4;                    // wave32s per block
constexpr int TPB   = WPB * 32;             // 128 threads
constexpr int CC    = 4 * C_;               // 256 cat channels
constexpr int KTOT  = CC * KS * KS;         // 2304 im2col K
constexpr int KCH   = KTOT / 32;            // 72 K-chunks
constexpr int PIX2  = 32;                   // pixels per K2 block (2 subtiles)
constexpr int TW    = PIX2 + 2;             // staged cat-tile width (halo)

// K1 x halo tile: rows h-5..h+5 (11 used, padded 16), cols w0-5..w0+20
// (26 used, padded 32) -> shift-only indexing, bf16 storage.
constexpr int XT_R = 16, XT_C = 32, XT_CH = XT_R * XT_C;   // 512/channel

constexpr int NBRANCH = N_ * 3 * C_ * H_ * W_;    // 12.58M floats
constexpr int AB1     = RT * 32 * 32;             // 110592 bf16 (gen_w swizzled)
constexpr int AB2     = 4 * KCH * 32 * 16;        // 147456 bf16 (fuse_w swizzled)

__device__ __forceinline__ unsigned short f2bf(float f) {
  unsigned u = __float_as_uint(f);
  u += 0x7FFFu + ((u >> 16) & 1u);          // round-to-nearest-even
  return (unsigned short)(u >> 16);
}
__device__ __forceinline__ float bf2f(unsigned short b) {
  return __uint_as_float((unsigned)b << 16);
}

// ISA 16-bit A/B fragment K index for element e (16x16x32 shapes):
// lanes 0-15 : e0..7 -> K0..7,  e8..15 -> K16..23
// lanes 16-31: e0..7 -> K8..15, e8..15 -> K24..31
__device__ __forceinline__ int frag_k(int lane, int e) {
  int hi = lane >> 4;
  int base = (e < 8) ? (hi ? 8 : 0) : (hi ? 24 : 16);
  return base + (e & 7);
}

// ---------------------------------------------------------------- prep ------
__global__ void __launch_bounds__(256)
prep_kernel(const float* __restrict__ gen_w, const float* __restrict__ fuse_w,
            unsigned short* __restrict__ abuf1, unsigned short* __restrict__ abuf2,
            int* __restrict__ rowlut, int* __restrict__ klut) {
  int t = blockIdx.x * 256 + threadIdx.x;
  if (t < AB1) {                             // gen_w -> [r][lane][32] frag order
    int r    = t >> 10;
    int rem  = t & 1023;
    int lane = rem >> 5;
    int e32  = rem & 31;
    int k    = frag_k(lane, e32 & 15) + (e32 >> 4) * 32;
    int mrow = r * 16 + (lane & 15);
    abuf1[t] = f2bf(gen_w[mrow * C_ + k]);
    return;
  }
  t -= AB1;
  if (t < AB2) {                             // fuse_w -> [octile][chunk][lane][16]
    int octile = t / (KCH * 512);
    int rem    = t % (KCH * 512);
    int chunk  = rem >> 9;
    int rem2   = rem & 511;
    int lane   = rem2 >> 4;
    int e      = rem2 & 15;
    int kk     = chunk * 32 + frag_k(lane, e);
    int oc     = octile * 16 + (lane & 15);
    abuf2[t]   = f2bf(fuse_w[oc * KTOT + kk]);
    return;
  }
  t -= AB2;
  if (t < ROWS) {                            // row -> {bc | x-tile LDS offset}
    int bi  = t / (C_ * 9);
    int rem = t % (C_ * 9);
    int ch  = rem / 9;
    int tap = rem % 9;
    int d   = 2 * bi + 1;                    // dilations 1,3,5
    int dh  = (tap / 3 - 1) * d;             // [-5..5]
    int dw  = (tap % 3 - 1) * d;
    int xoff = ch * XT_CH + (dh + 5) * XT_C + (dw + 5);
    rowlut[t] = ((bi * C_ + ch) << 16) | xoff;
    return;
  }
  t -= ROWS;
  if (t < KTOT) {                            // kk -> cat-tile LDS offset
    int ic  = t / 9;
    int tap = t % 9;
    klut[t] = ic * (3 * TW) + (tap / 3) * TW + (tap % 3);
  }
}

// ------------------------------------------------- K1: gen + depthwise ------
__global__ void __launch_bounds__(TPB)
gen_dw_kernel(const float* __restrict__ x, const float* __restrict__ y,
              const float* __restrict__ gen_b,
              const unsigned short* __restrict__ abuf1,
              const int* __restrict__ rowlut, float* __restrict__ branch) {
  __shared__ __attribute__((aligned(16))) float          acc[3 * C_ * PIX_T]; // 12 KB
  __shared__ __attribute__((aligned(16))) unsigned short xt[C_ * XT_CH];      // 64 KB
  __shared__ __attribute__((aligned(16))) unsigned short ybs[C_ * PIX_T];     // 2 KB
  __shared__ __attribute__((aligned(16))) int            rlut_s[ROWS];        // 6.9 KB
  __shared__ __attribute__((aligned(16))) float          bias_s[ROWS];        // 6.9 KB

  const int b  = blockIdx.x;
  const int n  = b / (H_ * (W_ / PIX_T));
  const int r0 = b % (H_ * (W_ / PIX_T));
  const int h  = r0 / (W_ / PIX_T);
  const int w0 = (r0 % (W_ / PIX_T)) * PIX_T;

  const int tid  = threadIdx.x;
  const int lane = tid & 31;
  const int wave = tid >> 5;

  for (int i = tid; i < 3 * C_ * PIX_T; i += TPB) acc[i] = 0.f;
  for (int i = tid; i < ROWS; i += TPB) {
    rlut_s[i] = rowlut[i];
    bias_s[i] = gen_b[i];
  }
  // stage x halo tile, zero-padded, shift-only indexing
  for (int i = tid; i < C_ * XT_CH; i += TPB) {
    int ch = i >> 9;
    int rr = (i >> 5) & 15;
    int cc = i & 31;
    int hh = h + rr - 5;
    int ww = w0 + cc - 5;
    float v = 0.f;
    if (rr < 11 && cc < 26 && (unsigned)hh < (unsigned)H_ && (unsigned)ww < (unsigned)W_)
      v = x[((n * C_ + ch) * H_ + hh) * W_ + ww];
    xt[i] = f2bf(v);
  }
  for (int i = tid; i < C_ * PIX_T; i += TPB) {
    int c  = i >> 4;
    int px = i & 15;
    ybs[i] = f2bf(y[((n * C_ + c) * H_ + h) * W_ + w0 + px]);
  }
  __syncthreads();

  // B fragments: y tile, K = input channel (two K=32 chunks cover C=64)
  const int ncol = lane & 15;
  v16us b0u, b1u;
#pragma unroll
  for (int e = 0; e < 16; ++e) {
    int k  = frag_k(lane, e);
    b0u[e] = ybs[k * PIX_T + ncol];
    b1u[e] = ybs[(k + 32) * PIX_T + ncol];
  }
  const v16bf B0 = __builtin_bit_cast(v16bf, b0u);
  const v16bf B1 = __builtin_bit_cast(v16bf, b1u);

  const int whalf = lane >> 4;

  for (int r = wave; r < RT; r += WPB) {
    const unsigned short* ap = &abuf1[(r * 32 + lane) * 32];
    __builtin_prefetch(ap + WPB * 1024, 0, 1);       // speculative, in-ws
    v16bf A0 = __builtin_bit_cast(v16bf, *(const v16us*)ap);
    v16bf A1 = __builtin_bit_cast(v16bf, *(const v16us*)(ap + 16));

    const int rowb = r * 16 + whalf * 8;             // D layout: VGPR v -> M=v/M=v+8
    // bias as C-init + LUTs, batched as ds_load_b128 clauses
    float4 bb0 = *(const float4*)&bias_s[rowb];
    float4 bb1 = *(const float4*)&bias_s[rowb + 4];
    int4   l0  = *(const int4*)&rlut_s[rowb];
    int4   l1  = *(const int4*)&rlut_s[rowb + 4];
    v8f c8 = {bb0.x, bb0.y, bb0.z, bb0.w, bb1.x, bb1.y, bb1.z, bb1.w};
    int luts[8] = {l0.x, l0.y, l0.z, l0.w, l1.x, l1.y, l1.z, l1.w};

    c8 = __builtin_amdgcn_wmma_f32_16x16x32_bf16(false, A0, false, B0,
                                                 (short)0, c8, false, false);
    c8 = __builtin_amdgcn_wmma_f32_16x16x32_bf16(false, A1, false, B1,
                                                 (short)0, c8, false, false);

    // phase 1: batched LDS gather of x (independent ds_load_u16 clause)
    float xv[8];
#pragma unroll
    for (int v = 0; v < 8; ++v)
      xv[v] = bf2f(xt[(luts[v] & 0xffff) + ncol]);

    // phase 2: LDS atomic accumulate (ds_add_f32)
#pragma unroll
    for (int v = 0; v < 8; ++v)
      atomicAdd(&acc[(luts[v] >> 16) * PIX_T + ncol], c8[v] * xv[v]);
  }
  __syncthreads();

  for (int i = tid; i < 3 * C_ * PIX_T; i += TPB) {
    int bc = i >> 4;
    int px = i & 15;
    branch[(((long)n * 3 * C_ + bc) * H_ + h) * W_ + w0 + px] = acc[i];
  }
}

// --------------------------------------------------- K2: fuse 3x3 conv ------
__global__ void __launch_bounds__(TPB)
fuse_conv_kernel(const float* __restrict__ x, const float* __restrict__ branch,
                 const unsigned short* __restrict__ abuf2,
                 const int* __restrict__ klut,
                 const float* __restrict__ fuse_b, float* __restrict__ out) {
  __shared__ __attribute__((aligned(16))) unsigned short tile[CC * 3 * TW]; // 52 KB
  __shared__ __attribute__((aligned(16))) int            klut_s[KTOT];      // 9.2 KB

  const int b  = blockIdx.x;
  const int n  = b / (H_ * (W_ / PIX2));
  const int r0 = b % (H_ * (W_ / PIX2));
  const int h  = r0 / (W_ / PIX2);
  const int w0 = (r0 % (W_ / PIX2)) * PIX2;

  const int tid  = threadIdx.x;
  const int lane = tid & 31;
  const int wave = tid >> 5;                       // = out-channel tile
  const int ncol = lane & 15;

  // stage cat = [x ; branch] tile (3 halo rows, zero-padded) as bf16
  for (int i = tid; i < CC * 3 * TW; i += TPB) {
    int ic  = i / (3 * TW);
    int rem = i % (3 * TW);
    int rr  = rem / TW;
    int cc  = rem % TW;
    int hh  = h + rr - 1;
    int ww  = w0 + cc - 1;
    float v = 0.f;
    if ((unsigned)hh < (unsigned)H_ && (unsigned)ww < (unsigned)W_)
      v = (ic < C_) ? x[((n * C_ + ic) * H_ + hh) * W_ + ww]
                    : branch[(((long)n * 3 * C_ + (ic - C_)) * H_ + hh) * W_ + ww];
    tile[i] = f2bf(v);
  }
  for (int i = tid; i < KTOT; i += TPB) klut_s[i] = klut[i];
  __syncthreads();

  const int kb = (lane >= 16) ? 8 : 0;             // frag_k low-run base

  v8f c0 = {}, c1 = {};
  for (int kt = 0; kt < KCH; ++kt) {
    const unsigned short* ap = &abuf2[((wave * KCH + kt) * 32 + lane) * 16];
    __builtin_prefetch(ap + 512, 0, 1);
    v16bf A = __builtin_bit_cast(v16bf, *(const v16us*)ap);

    // 16 LUT words as 4x ds_load_b128 (frag_k = two runs of 8 consecutive)
    const int* kl = &klut_s[kt * 32 + kb];
    int4 q0 = *(const int4*)kl;
    int4 q1 = *(const int4*)(kl + 4);
    int4 q2 = *(const int4*)(kl + 16);
    int4 q3 = *(const int4*)(kl + 20);
    int offs[16] = {q0.x, q0.y, q0.z, q0.w, q1.x, q1.y, q1.z, q1.w,
                    q2.x, q2.y, q2.z, q2.w, q3.x, q3.y, q3.z, q3.w};

    v16us bu0, bu1;
#pragma unroll
    for (int e = 0; e < 16; ++e) {
      int off = offs[e] + ncol;
      bu0[e] = tile[off];                          // subtile 0
      bu1[e] = tile[off + 16];                     // subtile 1 (shares A + LUT)
    }
    c0 = __builtin_amdgcn_wmma_f32_16x16x32_bf16(false, A, false,
           __builtin_bit_cast(v16bf, bu0), (short)0, c0, false, false);
    c1 = __builtin_amdgcn_wmma_f32_16x16x32_bf16(false, A, false,
           __builtin_bit_cast(v16bf, bu1), (short)0, c1, false, false);
  }

  const int whalf = lane >> 4;
#pragma unroll
  for (int v = 0; v < 8; ++v) {
    int oc = wave * 16 + v + whalf * 8;
    float bias = fuse_b[oc];
    long base = ((long)(n * C_ + oc) * H_ + h) * W_ + w0 + ncol;
    out[base]      = c0[v] + bias;
    out[base + 16] = c1[v] + bias;
  }
}

// ------------------------------------------------------------- launch -------
extern "C" void kernel_launch(void* const* d_in, const int* in_sizes, int n_in,
                              void* d_out, int out_size, void* d_ws, size_t ws_size,
                              hipStream_t stream) {
  (void)in_sizes; (void)n_in; (void)out_size; (void)ws_size;
  const float* x      = (const float*)d_in[0];
  const float* y      = (const float*)d_in[1];
  const float* gen_w  = (const float*)d_in[2];
  const float* gen_b  = (const float*)d_in[3];
  const float* fuse_w = (const float*)d_in[4];
  const float* fuse_b = (const float*)d_in[5];
  float* out = (float*)d_out;

  // workspace layout (32B-aligned): branch | abuf1 | abuf2 | rowlut | klut
  float*          branch = (float*)d_ws;                 // 50.33 MB
  unsigned short* abuf1  = (unsigned short*)(branch + NBRANCH);
  unsigned short* abuf2  = abuf1 + AB1;
  int*            rowlut = (int*)(abuf2 + AB2);
  int*            klut   = rowlut + ROWS;

  const int prep_elems = AB1 + AB2 + ROWS + KTOT;        // 262,080
  prep_kernel<<<(prep_elems + 255) / 256, 256, 0, stream>>>(
      gen_w, fuse_w, abuf1, abuf2, rowlut, klut);

  gen_dw_kernel<<<N_ * H_ * (W_ / PIX_T), TPB, 0, stream>>>(
      x, y, gen_b, abuf1, rowlut, branch);

  fuse_conv_kernel<<<N_ * H_ * (W_ / PIX2), TPB, 0, stream>>>(
      x, branch, abuf2, klut, fuse_b, out);
}
